// SwinV2Attention_71468255805916
// MI455X (gfx1250) — matrix operand order repaired and checked by reading, not compiled, not run on hardware
//
#include <hip/hip_runtime.h>
#include <hip/hip_bf16.h>
#include <math.h>

// ---------------------------------------------------------------------------
// SwinV2 window attention for gfx1250 (MI455X), wave32 + WMMA f16 16x16x32.
// ---------------------------------------------------------------------------

typedef _Float16 half_t;
typedef _Float16 v16h __attribute__((ext_vector_type(16)));
typedef float    v8f  __attribute__((ext_vector_type(8)));

#define LOG100F 4.6051702f

// ---- WMMA helpers ---------------------------------------------------------

__device__ __forceinline__ v8f zero8() {
    v8f z = {0.f, 0.f, 0.f, 0.f, 0.f, 0.f, 0.f, 0.f};
    return z;
}

__device__ __forceinline__ v8f wmma16(v16h a, v16h b, v8f c) {
    // D = A(16x32 f16) * B(32x16 f16) + C(16x16 f32)
    return __builtin_amdgcn_wmma_f32_16x16x32_f16(
        /*neg_a=*/false, a, /*neg_b=*/false, b,
        /*c_mod=*/(short)0, c, /*reuse_a=*/false, /*reuse_b=*/false);
}

// A-matrix fragment loader from row-major f16 storage.
// A layout used here (16-bit 16x32, contiguous k, not ISA table order): lane M = lane%16;
//   lanes 0-15 : halves 0..15 = K base+0..15
//   lanes16-31 : halves 0..15 = K base+16..31
__device__ __forceinline__ v16h load_a16(const half_t* base, int rowstride,
                                         int lane, int kbase) {
    int row = lane & 15, hi = lane >> 4;
    const half_t* p = base + row * rowstride + kbase + hi * 16;
    v16h a;
    ((uint4*)&a)[0] = *(const uint4*)(p);
    ((uint4*)&a)[1] = *(const uint4*)(p + 8);
    return a;
}

// B-matrix fragment loader where each B column n is a contiguous row of
// `base` (i.e. base holds Bᵀ row-major). B layout used here (32x16, contiguous k, same as A):
//   lane column N = lane%16; lanes 0-15 hold K 0..15, lanes 16-31 K 16..31.
__device__ __forceinline__ v16h load_b16(const half_t* base, int rowstride,
                                         int lane, int kbase) {
    int n = lane & 15, hi = lane >> 4;
    const half_t* p = base + n * rowstride + kbase + hi * 16;
    v16h b;
    ((uint4*)&b)[0] = *(const uint4*)(p);
    ((uint4*)&b)[1] = *(const uint4*)(p + 8);
    return b;
}

// Pre-swizzled global weight block: 512 halves, lane's 16 halves contiguous.
__device__ __forceinline__ v16h load_b_global(const half_t* blk, int lane) {
    const uint4* p = (const uint4*)(blk + lane * 16);
    v16h b;
    ((uint4*)&b)[0] = p[0];
    ((uint4*)&b)[1] = p[1];
    return b;
}

// ---- half-wave (16-lane) reductions: rows of a C fragment live in a fixed
// 16-lane group (lanes 0-15 hold M=r, lanes 16-31 hold M=r+8). ---------------

__device__ __forceinline__ float red_add16(float v) {
    v += __shfl_xor(v, 1, 32);
    v += __shfl_xor(v, 2, 32);
    v += __shfl_xor(v, 4, 32);
    v += __shfl_xor(v, 8, 32);
    return v;
}
__device__ __forceinline__ float red_max16(float v) {
    v = fmaxf(v, __shfl_xor(v, 1, 32));
    v = fmaxf(v, __shfl_xor(v, 2, 32));
    v = fmaxf(v, __shfl_xor(v, 4, 32));
    v = fmaxf(v, __shfl_xor(v, 8, 32));
    return v;
}

// ---------------------------------------------------------------------------
// Prep kernel A: f16-swizzle qkv_w [128,384] and proj_w [128,128] into WMMA
// B-fragment blocks [ntile][ktile][lane(32)][16 halves]; build fused qkv bias
// (q_bias, 0, v_bias) and clamped per-head temperature.
// ---------------------------------------------------------------------------
__global__ void prep_weights_kernel(const float* __restrict__ qkv_w,
                                    const float* __restrict__ proj_w,
                                    const float* __restrict__ q_bias,
                                    const float* __restrict__ v_bias,
                                    const float* __restrict__ logit_scale,
                                    half_t* __restrict__ wqkv,
                                    half_t* __restrict__ wproj,
                                    float* __restrict__ qkvb,
                                    float* __restrict__ scale4) {
    const int T1 = 24 * 4 * 512;   // qkv blocks (N=384 -> 24 ntiles, K=128 -> 4 kt)
    const int T2 = 8 * 4 * 512;    // proj blocks (N=128 -> 8 ntiles)
    const int TT = T1 + T2 + 384 + 4;
    for (int t = blockIdx.x * blockDim.x + threadIdx.x; t < TT;
         t += gridDim.x * blockDim.x) {
        if (t < T1) {
            int blk = t >> 9, l = (t >> 4) & 31, p = t & 15;
            int nt = blk >> 2, kt = blk & 3;
            int n = nt * 16 + (l & 15);
            int k = kt * 32 + ((l >> 4) * 16) + p;
            wqkv[t] = (half_t)qkv_w[k * 384 + n];
        } else if (t < T1 + T2) {
            int u = t - T1;
            int blk = u >> 9, l = (u >> 4) & 31, p = u & 15;
            int nt = blk >> 2, kt = blk & 3;
            int n = nt * 16 + (l & 15);
            int k = kt * 32 + ((l >> 4) * 16) + p;
            wproj[u] = (half_t)proj_w[k * 128 + n];
        } else if (t < T1 + T2 + 384) {
            int c = t - T1 - T2;
            qkvb[c] = (c < 128) ? q_bias[c] : (c < 256 ? 0.f : v_bias[c - 256]);
        } else {
            int h = t - T1 - T2 - 384;
            scale4[h] = __expf(fminf(logit_scale[h], LOG100F));
        }
    }
}

// ---------------------------------------------------------------------------
// Prep kernel B: CPB MLP -> bias4[h][i][j] = 16*sigmoid(MLP(table)[idx(i,j)][h])
// ---------------------------------------------------------------------------
__device__ __forceinline__ float cpb_coord(int r) {
    float v = (float)r * (8.0f / 7.0f);
    float lv = log2f(fabsf(v) + 1.0f) * (1.0f / log2f(8.0f));
    return (v < 0.f) ? -lv : lv;
}

__global__ void prep_bias_kernel(const float* __restrict__ w1,   // [2,512]
                                 const float* __restrict__ b1,   // [512]
                                 const float* __restrict__ w2,   // [512,4]
                                 float* __restrict__ bias4) {    // [4,64,64]
    __shared__ float tab[225 * 4];
    const int tid = threadIdx.x;
    if (tid < 225) {
        float t0 = cpb_coord(tid / 15 - 7);
        float t1 = cpb_coord(tid % 15 - 7);
        float a0 = 0.f, a1 = 0.f, a2 = 0.f, a3 = 0.f;
        for (int c = 0; c < 512; ++c) {
            float h = fmaxf(t0 * w1[c] + t1 * w1[512 + c] + b1[c], 0.f);
            a0 += h * w2[c * 4 + 0];
            a1 += h * w2[c * 4 + 1];
            a2 += h * w2[c * 4 + 2];
            a3 += h * w2[c * 4 + 3];
        }
        tab[tid * 4 + 0] = a0;
        tab[tid * 4 + 1] = a1;
        tab[tid * 4 + 2] = a2;
        tab[tid * 4 + 3] = a3;
    }
    __syncthreads();
    for (int t = tid; t < 4 * 64 * 64; t += blockDim.x) {
        int h = t >> 12, i = (t >> 6) & 63, j = t & 63;
        int idx = ((i >> 3) - (j >> 3) + 7) * 15 + ((i & 7) - (j & 7) + 7);
        float bt = tab[idx * 4 + h];
        bias4[t] = 16.f / (1.f + __expf(-bt));
    }
}

// ---------------------------------------------------------------------------
// Main kernel: one workgroup (8 waves) per window.
//   LDS (64 KB):
//     [ 0,16K)  s_x  : x-tile f16 [64][128]   -> reused as P staging (8x2KB)
//     [16,32K)  s_q  : q̂  f16 [4][64][32]     -> reused as attn-out [64][128]
//     [32,48K)  s_k  : k̂  f16 [4][64][32]
//     [48,64K)  s_vT : Vᵀ f16 [4][32][64]
// ---------------------------------------------------------------------------
__global__ __launch_bounds__(256) void swin_attn_kernel(
    const float* __restrict__ x, const float* __restrict__ mask,
    const half_t* __restrict__ wqkv, const half_t* __restrict__ wproj,
    const float* __restrict__ qkvb, const float* __restrict__ scale4,
    const float* __restrict__ bias4, const float* __restrict__ proj_b,
    float* __restrict__ out) {
    __shared__ __align__(16) half_t smem[32768];
    half_t* s_x  = smem;
    half_t* s_q  = smem + 8192;
    half_t* s_k  = smem + 16384;
    half_t* s_vT = smem + 24576;

    const int tid  = threadIdx.x;
    const int lane = tid & 31, wave = tid >> 5;
    const int ln = lane & 15, hi = lane >> 4;
    const int win = blockIdx.x;
    const int b = win >> 8, wflat = win & 255;
    const int whi = wflat >> 4, wwi = wflat & 15;

    // Warm L2 with the (pre-swizzled) weights — gfx1250 global_prefetch path.
    for (int pfo = tid * 128; pfo < (98304 + 32768); pfo += 256 * 128)
        __builtin_prefetch((const char*)wqkv + pfo, 0, 0);

    // ---- load 64x128 fp32 window tile -> f16 LDS --------------------------
    for (int i = tid; i < 64 * 32; i += 256) {
        int r = i >> 5, c4 = i & 31;
        int y = whi * 8 + (r >> 3), xc = wwi * 8 + (r & 7);
        const float4 v =
            ((const float4*)(x + (((size_t)(b * 128 + y)) * 128 + xc) * 128))[c4];
        half_t* d = s_x + r * 128 + c4 * 4;
        d[0] = (half_t)v.x; d[1] = (half_t)v.y;
        d[2] = (half_t)v.z; d[3] = (half_t)v.w;
    }
    __syncthreads();

    // ---- QKV GEMM [64,128]x[128,384] + bias + cosine norm + temperature ---
    {
        int mi = wave & 3, ng = wave >> 2;     // M-tile, N-group (12 ntiles each)
        v16h afr[4];
#pragma unroll
        for (int kt = 0; kt < 4; ++kt)
            afr[kt] = load_a16(s_x + mi * 16 * 128, 128, lane, kt * 32);

#pragma unroll
        for (int np = 0; np < 6; ++np) {       // head-aligned 32-col pairs
            int nt0 = ng * 12 + np * 2;
            v8f c0 = zero8(), c1 = zero8();
#pragma unroll
            for (int kt = 0; kt < 4; ++kt) {
                v16h b0 = load_b_global(wqkv + (((nt0 + 0) * 4 + kt) << 9), lane);
                v16h b1 = load_b_global(wqkv + (((nt0 + 1) * 4 + kt) << 9), lane);
                c0 = wmma16(afr[kt], b0, c0);
                c1 = wmma16(afr[kt], b1, c1);
            }
            int col0 = nt0 * 16;
            float bb0 = qkvb[col0 + ln], bb1 = qkvb[col0 + 16 + ln];
#pragma unroll
            for (int r = 0; r < 8; ++r) { c0[r] += bb0; c1[r] += bb1; }

            int sec = col0 >> 7;               // 0=q, 1=k, 2=v
            int hh = (col0 & 127) >> 5;        // head
            if (sec < 2) {
                float sc = (sec == 0) ? scale4[hh] : 1.0f;
                half_t* dst = (sec == 0 ? s_q : s_k) + hh * 2048;
#pragma unroll
                for (int r = 0; r < 8; ++r) {
                    float ss = red_add16(c0[r] * c0[r] + c1[r] * c1[r]);
                    float inv = sc / fmaxf(sqrtf(ss), 1e-12f);
                    int row = mi * 16 + r + hi * 8;
                    dst[row * 32 + ln]      = (half_t)(c0[r] * inv);
                    dst[row * 32 + 16 + ln] = (half_t)(c1[r] * inv);
                }
            } else {                            // V stored transposed [d][k]
                half_t* dst = s_vT + hh * 2048;
#pragma unroll
                for (int r = 0; r < 8; ++r) {
                    int row = mi * 16 + r + hi * 8;
                    dst[ln * 64 + row]        = (half_t)c0[r];
                    dst[(16 + ln) * 64 + row] = (half_t)c1[r];
                }
            }
        }
    }
    __syncthreads();

    // ---- attention: 2 waves per head, 32 rows each -------------------------
    {
        int h = wave >> 1, mhalf = wave & 1;
        const half_t* qh = s_q + h * 2048;
        const half_t* kh = s_k + h * 2048;
        const half_t* vh = s_vT + h * 2048;

        v8f S[2][4];
#pragma unroll
        for (int mt2 = 0; mt2 < 2; ++mt2) {
            int mt = mhalf * 2 + mt2;
            v16h aq = load_a16(qh + mt * 16 * 32, 32, lane, 0);
#pragma unroll
            for (int nt = 0; nt < 4; ++nt) {
                v16h bk = load_b16(kh + nt * 16 * 32, 32, lane, 0);
                S[mt2][nt] = wmma16(aq, bk, zero8());
            }
        }

        // + CPB bias + window mask, register softmax (16-lane reductions)
        const float* bh = bias4 + h * 4096;
        const float* mw = mask + (size_t)wflat * 4096;
#pragma unroll
        for (int mt2 = 0; mt2 < 2; ++mt2) {
#pragma unroll
            for (int r = 0; r < 8; ++r) {
                int row = (mhalf * 2 + mt2) * 16 + r + hi * 8;
                float vals[4];
                float mx = -3.0e38f;
#pragma unroll
                for (int nt = 0; nt < 4; ++nt) {
                    float t = S[mt2][nt][r] + bh[row * 64 + nt * 16 + ln] +
                              mw[row * 64 + nt * 16 + ln];
                    vals[nt] = t;
                    mx = fmaxf(mx, t);
                }
                mx = red_max16(mx);
                float sum = 0.f;
#pragma unroll
                for (int nt = 0; nt < 4; ++nt) {
                    vals[nt] = __expf(vals[nt] - mx);
                    sum += vals[nt];
                }
                sum = red_add16(sum);
                float inv = 1.0f / sum;
#pragma unroll
                for (int nt = 0; nt < 4; ++nt) S[mt2][nt][r] = vals[nt] * inv;
            }
        }
        __syncthreads();   // every wave finished reading q/k (s_q reused below)

        // ---- out = P @ V ; stage P per-wave (2 KB slice of s_x) -----------
        half_t* ps  = s_x + wave * 1024;       // [16][64]
        half_t* s_o = s_q;                     // attn-out [64][128]
#pragma unroll
        for (int mt2 = 0; mt2 < 2; ++mt2) {
            int mt = mhalf * 2 + mt2;
#pragma unroll
            for (int r = 0; r < 8; ++r) {
                int rl = r + hi * 8;
#pragma unroll
                for (int nt = 0; nt < 4; ++nt)
                    ps[rl * 64 + nt * 16 + ln] = (half_t)S[mt2][nt][r];
            }
            v16h ap0 = load_a16(ps, 64, lane, 0);
            v16h ap1 = load_a16(ps, 64, lane, 32);
#pragma unroll
            for (int nt = 0; nt < 2; ++nt) {
                v8f co = zero8();
                co = wmma16(ap0, load_b16(vh + nt * 16 * 64, 64, lane, 0), co);
                co = wmma16(ap1, load_b16(vh + nt * 16 * 64, 64, lane, 32), co);
#pragma unroll
                for (int r = 0; r < 8; ++r) {
                    int row = mt * 16 + r + hi * 8;
                    s_o[row * 128 + h * 32 + nt * 16 + ln] = (half_t)co[r];
                }
            }
        }
    }
    __syncthreads();

    // ---- projection [64,128]x[128,128] + bias -> global fp32 --------------
    {
        half_t* s_o = s_q;
        int mi = wave & 3, ngp = wave >> 2;
        v16h ao[4];
#pragma unroll
        for (int kt = 0; kt < 4; ++kt)
            ao[kt] = load_a16(s_o + mi * 16 * 128, 128, lane, kt * 32);
#pragma unroll
        for (int nti = 0; nti < 4; ++nti) {
            int nt = ngp * 4 + nti;
            v8f c = zero8();
#pragma unroll
            for (int kt = 0; kt < 4; ++kt)
                c = wmma16(ao[kt], load_b_global(wproj + ((nt * 4 + kt) << 9), lane), c);
            float pb = proj_b[nt * 16 + ln];
#pragma unroll
            for (int r = 0; r < 8; ++r) {
                int row = mi * 16 + r + hi * 8;
                int y = whi * 8 + (row >> 3), xc = wwi * 8 + (row & 7);
                out[(((size_t)(b * 128 + y)) * 128 + xc) * 128 + nt * 16 + ln] =
                    c[r] + pb;
            }
        }
    }
}

// ---------------------------------------------------------------------------

extern "C" void kernel_launch(void* const* d_in, const int* in_sizes, int n_in,
                              void* d_out, int out_size, void* d_ws, size_t ws_size,
                              hipStream_t stream) {
    const float* x           = (const float*)d_in[0];
    const float* mask        = (const float*)d_in[1];
    const float* qkv_w       = (const float*)d_in[2];
    const float* q_bias      = (const float*)d_in[3];
    const float* v_bias      = (const float*)d_in[4];
    const float* logit_scale = (const float*)d_in[5];
    const float* cpb_w1      = (const float*)d_in[6];
    const float* cpb_b1      = (const float*)d_in[7];
    const float* cpb_w2      = (const float*)d_in[8];
    const float* proj_w      = (const float*)d_in[9];
    const float* proj_b      = (const float*)d_in[10];
    float* out = (float*)d_out;

    char* ws = (char*)d_ws;
    half_t* wqkv  = (half_t*)(ws);            //  98304 B
    half_t* wproj = (half_t*)(ws + 98304);    //  32768 B
    float*  qkvb  = (float*)(ws + 131072);    //   1536 B
    float*  scale = (float*)(ws + 132608);    //     16 B
    float*  bias4 = (float*)(ws + 132624);    //  65536 B

    prep_weights_kernel<<<64, 256, 0, stream>>>(qkv_w, proj_w, q_bias, v_bias,
                                                logit_scale, wqkv, wproj, qkvb,
                                                scale);
    prep_bias_kernel<<<1, 256, 0, stream>>>(cpb_w1, cpb_b1, cpb_w2, bias4);

    swin_attn_kernel<<<4096, 256, 0, stream>>>(x, mask, wqkv, wproj, qkvb,
                                               scale, bias4, proj_b, out);
}
